// TransformerBlock_1125281432245
// MI455X (gfx1250) — matrix operand order repaired and checked by reading, not compiled, not run on hardware
//
#include <hip/hip_runtime.h>
#include <cstdint>
#include <cstddef>

#define D_MODEL 1024
#define NHEAD   16
#define HEADD   64
#define SEQ     2048
#define BATCH   2
#define FFDIM   4096
#define NTOK    (BATCH * SEQ)   /* 4096 token rows */

typedef __bf16 bf16;
typedef __attribute__((ext_vector_type(16))) __bf16 v16bf;
typedef __attribute__((ext_vector_type(8)))  __bf16 v8bf;
typedef __attribute__((ext_vector_type(8)))  float  v8f;

static __device__ __forceinline__ bf16 f2bf(float f) {
  unsigned u = __builtin_bit_cast(unsigned, f);
  unsigned r = u + 0x7fffu + ((u >> 16) & 1u);   // round-to-nearest-even
  unsigned short h = (unsigned short)(r >> 16);
  return __builtin_bit_cast(bf16, h);
}

static __device__ __forceinline__ v16bf pack16(v8bf lo, v8bf hi) {
  v16bf t;
#pragma unroll
  for (int e = 0; e < 8; ++e) { t[e] = lo[e]; t[e + 8] = hi[e]; }
  return t;
}

static __device__ __forceinline__ v8f wmma_bf16(v16bf a, v16bf b, v8f c) {
  // D = A(16x32 bf16) * B(32x16 bf16) + C(16x16 f32)
  return __builtin_amdgcn_wmma_f32_16x16x32_bf16(false, a, false, b, (short)0, c,
                                                 false, false);
}

// async global -> LDS, 16 bytes per lane, tracked by ASYNCcnt
static __device__ __forceinline__ void async_ld16(unsigned lds_off, const bf16* gp) {
  asm volatile("global_load_async_to_lds_b128 %0, %1, off"
               :: "v"(lds_off), "v"(gp)
               : "memory");
}

static __device__ __forceinline__ float gelu_tanh(float x) {
  const float c = 0.79788456080286535588f;  // sqrt(2/pi)
  return 0.5f * x * (1.0f + tanhf(c * (x + 0.044715f * x * x * x)));
}

// ---------------------------------------------------------------- cvt fp32->bf16
__global__ __launch_bounds__(256) void cvt_f32_bf16(const float* __restrict__ src,
                                                    bf16* __restrict__ dst, int n) {
  int i = blockIdx.x * 256 + threadIdx.x;
  if (i < n) dst[i] = f2bf(src[i]);
}

// ---------------------------------------------------------------- LayerNorm (torch-faithful)
__global__ __launch_bounds__(256) void layernorm_bf16(const float* __restrict__ x,
                                                      const float* __restrict__ scale,
                                                      const float* __restrict__ shift,
                                                      bf16* __restrict__ out) {
  __shared__ float red[512];
  const int row = blockIdx.x;
  const int tid = threadIdx.x;
  const float* xr = x + (size_t)row * D_MODEL;
  float v[4];
  float s = 0.f, s2 = 0.f;
#pragma unroll
  for (int i = 0; i < 4; ++i) {
    v[i] = xr[tid + 256 * i];
    s += v[i];
    s2 += v[i] * v[i];
  }
  red[tid] = s;
  red[256 + tid] = s2;
  __syncthreads();
  for (int off = 128; off > 0; off >>= 1) {
    if (tid < off) {
      red[tid] += red[tid + off];
      red[256 + tid] += red[256 + tid + off];
    }
    __syncthreads();
  }
  const float sum = red[0], sumsq = red[256];
  const float mean = sum * (1.0f / D_MODEL);
  const float var = (sumsq - sum * sum * (1.0f / D_MODEL)) * (1.0f / (D_MODEL - 1));
  const float inv = 1.0f / (sqrtf(var) + 1e-5f);
  bf16* orow = out + (size_t)row * D_MODEL;
#pragma unroll
  for (int i = 0; i < 4; ++i) {
    int d = tid + 256 * i;
    orow[d] = f2bf(shift[d] + scale[d] * (v[i] - mean) * inv);
  }
}

// ---------------------------------------------------------------- WMMA GEMM: C = A[M,K] * W[N,K]^T
// Async global->LDS, double-buffered, K=64 per stage, 128x128 block tile; 8 waves, each 32x64.
#define GF_GELU 1
#define GF_BF16 2
#define LDSW 72   /* bf16 per staged row: 64 data + 8 pad (144B rows -> bank-conflict-free) */

__global__ __launch_bounds__(256) void gemm_wmma_bf16(const bf16* __restrict__ A,
                                                      const bf16* __restrict__ W,
                                                      void* out, const float* __restrict__ bias,
                                                      const float* resid, int M, int N, int K,
                                                      int flags) {
  __shared__ __align__(128) bf16 smem[2][2][128 * LDSW];  // 73728 B: [buf][A/B][row*LDSW]
  const int tid = threadIdx.x;
  const int lane = tid & 31;
  const int wv = tid >> 5;
  const int lh = lane & 15, hic = lane >> 4;
  const int wm = wv & 3, wn = wv >> 2;
  const int m0 = blockIdx.y * 128;
  const int n0 = blockIdx.x * 128;
  (void)M;

  // per-stage tile = 128 rows x 128 bytes = 1024 16B chunks per matrix; 4 per thread each
  const bf16* gA[4];
  const bf16* gB[4];
  unsigned lofs[4];
#pragma unroll
  for (int i = 0; i < 4; ++i) {
    const int c = tid * 4 + i;
    const int row = c >> 3, part = c & 7;
    gA[i] = A + (size_t)(m0 + row) * K + part * 8;
    gB[i] = W + (size_t)(n0 + row) * K + part * 8;
    lofs[i] = (unsigned)(row * LDSW * 2 + part * 16);
  }
  const unsigned sbase = (unsigned)(uintptr_t)(&smem[0][0][0]);
  const unsigned bufBytes = 2u * 128 * LDSW * 2;   // 36864
  const unsigned halfBytes = 128u * LDSW * 2;      // 18432

  auto issue = [&](int buf, int kk) {
    const unsigned ab = sbase + (unsigned)buf * bufBytes;
#pragma unroll
    for (int i = 0; i < 4; ++i) async_ld16(ab + lofs[i], gA[i] + kk);
#pragma unroll
    for (int i = 0; i < 4; ++i) async_ld16(ab + halfBytes + lofs[i], gB[i] + kk);
  };

  v8f acc[2][4];
#pragma unroll
  for (int ms = 0; ms < 2; ++ms)
#pragma unroll
    for (int t = 0; t < 4; ++t)
#pragma unroll
      for (int r = 0; r < 8; ++r) acc[ms][t][r] = 0.f;

  const int nk = K >> 6;   // stages of K=64
  issue(0, 0);
  for (int j = 0; j < nk; ++j) {
    asm volatile("s_wait_asynccnt 0" ::: "memory");
    __syncthreads();
    if (j + 1 < nk) issue((j + 1) & 1, (j + 1) * 64);

    const bf16* sA = &smem[j & 1][0][0];
    const bf16* sB = &smem[j & 1][1][0];

#pragma unroll
    for (int ks = 0; ks < 2; ++ks) {
      const int ko = ks * 32;
      // all 12 fragment loads of this k-slice issued up front
      const bf16* a0p = sA + (wm * 32 + lh) * LDSW + ko + hic * 8;
      const bf16* a1p = a0p + 16 * LDSW;
      const bf16* bp = sB + (wn * 64 + lh) * LDSW + ko + hic * 8;
      v16bf af0 = pack16(*(const v8bf*)a0p, *(const v8bf*)(a0p + 16));
      v16bf af1 = pack16(*(const v8bf*)a1p, *(const v8bf*)(a1p + 16));
      v16bf bv0 = pack16(*(const v8bf*)(bp),             *(const v8bf*)(bp + 16));
      v16bf bv1 = pack16(*(const v8bf*)(bp + 16 * LDSW), *(const v8bf*)(bp + 16 * LDSW + 16));
      v16bf bv2 = pack16(*(const v8bf*)(bp + 32 * LDSW), *(const v8bf*)(bp + 32 * LDSW + 16));
      v16bf bv3 = pack16(*(const v8bf*)(bp + 48 * LDSW), *(const v8bf*)(bp + 48 * LDSW + 16));
      acc[0][0] = wmma_bf16(af0, bv0, acc[0][0]);
      acc[1][0] = wmma_bf16(af1, bv0, acc[1][0]);
      acc[0][1] = wmma_bf16(af0, bv1, acc[0][1]);
      acc[1][1] = wmma_bf16(af1, bv1, acc[1][1]);
      acc[0][2] = wmma_bf16(af0, bv2, acc[0][2]);
      acc[1][2] = wmma_bf16(af1, bv2, acc[1][2]);
      acc[0][3] = wmma_bf16(af0, bv3, acc[0][3]);
      acc[1][3] = wmma_bf16(af1, bv3, acc[1][3]);
    }
  }

#pragma unroll
  for (int ms = 0; ms < 2; ++ms)
#pragma unroll
    for (int t = 0; t < 4; ++t)
#pragma unroll
      for (int r = 0; r < 8; ++r) {
        const int row = m0 + wm * 32 + ms * 16 + r + hic * 8;
        const int col = n0 + wn * 64 + t * 16 + lh;
        float vv = acc[ms][t][r];
        if (bias) vv += bias[col];
        if (flags & GF_GELU) vv = gelu_tanh(vv);
        if (resid) vv += resid[(size_t)row * N + col];
        if (flags & GF_BF16)
          ((bf16*)out)[(size_t)row * N + col] = f2bf(vv);
        else
          ((float*)out)[(size_t)row * N + col] = vv;
      }
}

// ---------------------------------------------------------------- qkv -> q[B,H,S,64], k[B,H,S,64], v^T[B,H,64,S]
__global__ __launch_bounds__(256) void qkv_rearrange(const bf16* __restrict__ qkv,
                                                     bf16* __restrict__ q,
                                                     bf16* __restrict__ k,
                                                     bf16* __restrict__ vt) {
  int idx = blockIdx.x * 256 + threadIdx.x;   // over NTOK * D_MODEL
  int d = idx & (D_MODEL - 1);
  int tok = idx >> 10;        // b*SEQ + s
  int b = tok >> 11;          // / SEQ
  int s = tok & (SEQ - 1);
  int h = d >> 6, hd = d & 63;
  size_t src = (size_t)tok * (3 * D_MODEL);
  size_t bh = (size_t)(b * NHEAD + h);
  q[(bh * SEQ + s) * HEADD + hd] = qkv[src + d];
  k[(bh * SEQ + s) * HEADD + hd] = qkv[src + D_MODEL + d];
  vt[(bh * HEADD + hd) * SEQ + s] = qkv[src + 2 * D_MODEL + d];
}

// ---------------------------------------------------------------- flash attention (causal), 1 wave = 16 queries
__global__ __launch_bounds__(128) void flash_attn(const bf16* __restrict__ Q,
                                                  const bf16* __restrict__ Kk,
                                                  const bf16* __restrict__ Vt,
                                                  bf16* __restrict__ O) {
  __shared__ __align__(16) bf16 plds[4 * 16 * 32];   // per-wave 16x32 P tile
  const int lane = threadIdx.x & 31;
  const int wv = threadIdx.x >> 5;
  const int lh = lane & 15, hic = lane >> 4;
  const int tile = blockIdx.x * 4 + wv;
  const int TQ = SEQ / 16;
  const int bh = tile / TQ;
  const int q0 = (tile % TQ) * 16;
  const int bb = bh / NHEAD, hh = bh % NHEAD;

  const bf16* Qp = Q + ((size_t)bh * SEQ + (q0 + lh)) * HEADD;
  const bf16* Kbase = Kk + (size_t)bh * SEQ * HEADD;
  const bf16* Vbase = Vt + (size_t)bh * HEADD * SEQ;

  v16bf qa[2];
#pragma unroll
  for (int s2 = 0; s2 < 2; ++s2) {
    v8bf lo = *(const v8bf*)(Qp + s2 * 32 + hic * 16);
    v8bf hi8 = *(const v8bf*)(Qp + s2 * 32 + hic * 16 + 8);
    qa[s2] = pack16(lo, hi8);
  }

  float mrow[8], lrow[8];
  v8f o[4];
#pragma unroll
  for (int r = 0; r < 8; ++r) { mrow[r] = -1e30f; lrow[r] = 0.f; }
#pragma unroll
  for (int t = 0; t < 4; ++t)
#pragma unroll
    for (int r = 0; r < 8; ++r) o[t][r] = 0.f;

  bf16* pw = plds + wv * 512;
  const int nb = (q0 + 47) >> 5;   // key blocks of 32 covering keys <= q0+15
  for (int j = 0; j < nb; ++j) {
    const int k0 = j * 32;
    v8f c[2];
#pragma unroll
    for (int t = 0; t < 2; ++t)
#pragma unroll
      for (int r = 0; r < 8; ++r) c[t][r] = 0.f;
#pragma unroll
    for (int t = 0; t < 2; ++t) {
      const bf16* Kp = Kbase + (size_t)(k0 + t * 16 + lh) * HEADD + hic * 16;
      v16bf bk0 = *(const v16bf*)(Kp);
      v16bf bk1 = *(const v16bf*)(Kp + 32);
      c[t] = wmma_bf16(qa[0], bk0, c[t]);
      c[t] = wmma_bf16(qa[1], bk1, c[t]);
    }
    // scale + causal mask + per-row max (rows live in accumulator elements)
    float rm[8];
#pragma unroll
    for (int r = 0; r < 8; ++r) {
      const int qrow = q0 + r + hic * 8;
      float s0 = c[0][r] * 0.125f;               // 1/sqrt(64)
      float s1 = c[1][r] * 0.125f;
      if (k0 + lh > qrow) s0 = -1e30f;
      if (k0 + 16 + lh > qrow) s1 = -1e30f;
      c[0][r] = s0;
      c[1][r] = s1;
      rm[r] = fmaxf(s0, s1);
    }
#pragma unroll
    for (int msk = 1; msk < 16; msk <<= 1)
#pragma unroll
      for (int r = 0; r < 8; ++r) rm[r] = fmaxf(rm[r], __shfl_xor(rm[r], msk, 32));

    float alpha[8], rs[8];
#pragma unroll
    for (int r = 0; r < 8; ++r) {
      float mnew = fmaxf(mrow[r], rm[r]);
      alpha[r] = __expf(mrow[r] - mnew);
      mrow[r] = mnew;
      float p0 = __expf(c[0][r] - mnew);
      float p1 = __expf(c[1][r] - mnew);
      c[0][r] = p0;
      c[1][r] = p1;
      rs[r] = p0 + p1;
    }
#pragma unroll
    for (int msk = 1; msk < 16; msk <<= 1)
#pragma unroll
      for (int r = 0; r < 8; ++r) rs[r] += __shfl_xor(rs[r], msk, 32);
#pragma unroll
    for (int r = 0; r < 8; ++r) lrow[r] = lrow[r] * alpha[r] + rs[r];
#pragma unroll
    for (int t = 0; t < 4; ++t)
#pragma unroll
      for (int r = 0; r < 8; ++r) o[t][r] *= alpha[r];

    // P (C-layout) -> LDS row-major 16x32 -> A fragment. LDS is in-order per wave.
#pragma unroll
    for (int t = 0; t < 2; ++t)
#pragma unroll
      for (int r = 0; r < 8; ++r)
        pw[(r + hic * 8) * 32 + t * 16 + lh] = f2bf(c[t][r]);
    asm volatile("s_wait_dscnt 0" ::: "memory");
    v8bf plo = *(const v8bf*)(pw + lh * 32 + hic * 16);
    v8bf phi = *(const v8bf*)(pw + lh * 32 + hic * 16 + 8);
    v16bf pa = pack16(plo, phi);
    asm volatile("s_wait_dscnt 0" ::: "memory");

#pragma unroll
    for (int t = 0; t < 4; ++t) {
      const bf16* Vp = Vbase + (size_t)(t * 16 + lh) * SEQ + k0 + hic * 16;
      v16bf bv = *(const v16bf*)(Vp);
      o[t] = wmma_bf16(pa, bv, o[t]);
    }
  }

#pragma unroll
  for (int t = 0; t < 4; ++t)
#pragma unroll
    for (int r = 0; r < 8; ++r) {
      const int srow = q0 + r + hic * 8;
      const int col = hh * HEADD + t * 16 + lh;
      O[((size_t)bb * SEQ + srow) * D_MODEL + col] = f2bf(o[t][r] / lrow[r]);
    }
}

// ---------------------------------------------------------------- launch
extern "C" void kernel_launch(void* const* d_in, const int* in_sizes, int n_in,
                              void* d_out, int out_size, void* d_ws, size_t ws_size,
                              hipStream_t stream) {
  (void)in_sizes; (void)n_in; (void)out_size; (void)ws_size;
  const float* x      = (const float*)d_in[0];
  const float* scale1 = (const float*)d_in[1];
  const float* shift1 = (const float*)d_in[2];
  const float* Wqkv   = (const float*)d_in[3];
  const float* Wo_w   = (const float*)d_in[4];
  const float* Wo_b   = (const float*)d_in[5];
  const float* scale2 = (const float*)d_in[6];
  const float* shift2 = (const float*)d_in[7];
  const float* W1     = (const float*)d_in[8];
  const float* b1     = (const float*)d_in[9];
  const float* W2     = (const float*)d_in[10];
  const float* b2     = (const float*)d_in[11];
  float* out = (float*)d_out;

  char* ws = (char*)d_ws;
  size_t off = 0;
  auto take = [&](size_t elems) {
    bf16* p = (bf16*)(ws + off);
    off += elems * sizeof(bf16);
    return p;
  };
  bf16* wqkv_b = take((size_t)3 * D_MODEL * D_MODEL);   // 3072x1024
  bf16* wo_b16 = take((size_t)D_MODEL * D_MODEL);       // 1024x1024
  bf16* w1_b16 = take((size_t)FFDIM * D_MODEL);         // 4096x1024
  bf16* w2_b16 = take((size_t)D_MODEL * FFDIM);         // 1024x4096
  bf16* hbf    = take((size_t)NTOK * D_MODEL);
  bf16* qkvbf  = take((size_t)NTOK * 3 * D_MODEL);
  bf16* qbf    = take((size_t)NTOK * D_MODEL);
  bf16* kbf    = take((size_t)NTOK * D_MODEL);
  bf16* vtbf   = take((size_t)NTOK * D_MODEL);
  bf16* attnbf = take((size_t)NTOK * D_MODEL);
  bf16* h2bf   = take((size_t)NTOK * D_MODEL);
  bf16* mbf    = take((size_t)NTOK * FFDIM);

  // weights -> bf16 (per launch; capture-safe, deterministic)
  cvt_f32_bf16<<<(3 * D_MODEL * D_MODEL) / 256, 256, 0, stream>>>(Wqkv, wqkv_b, 3 * D_MODEL * D_MODEL);
  cvt_f32_bf16<<<(D_MODEL * D_MODEL) / 256, 256, 0, stream>>>(Wo_w, wo_b16, D_MODEL * D_MODEL);
  cvt_f32_bf16<<<(FFDIM * D_MODEL) / 256, 256, 0, stream>>>(W1, w1_b16, FFDIM * D_MODEL);
  cvt_f32_bf16<<<(D_MODEL * FFDIM) / 256, 256, 0, stream>>>(W2, w2_b16, D_MODEL * FFDIM);

  // LN1 -> h (bf16)
  layernorm_bf16<<<NTOK, 256, 0, stream>>>(x, scale1, shift1, hbf);
  // qkv = h * Wqkv^T  (bf16 out)
  gemm_wmma_bf16<<<dim3(3 * D_MODEL / 128, NTOK / 128), 256, 0, stream>>>(
      hbf, wqkv_b, qkvbf, nullptr, nullptr, NTOK, 3 * D_MODEL, D_MODEL, GF_BF16);
  // split heads; v transposed for contiguous P*V B-fragments
  qkv_rearrange<<<NTOK * D_MODEL / 256, 256, 0, stream>>>(qkvbf, qbf, kbf, vtbf);
  // causal flash attention -> attn (bf16, [B,S,D])
  flash_attn<<<BATCH * NHEAD * (SEQ / 16) / 4, 128, 0, stream>>>(qbf, kbf, vtbf, attnbf);
  // x1 = x + attn*Wo^T + Wo_b  (fp32 into d_out)
  gemm_wmma_bf16<<<dim3(D_MODEL / 128, NTOK / 128), 256, 0, stream>>>(
      attnbf, wo_b16, (void*)out, Wo_b, x, NTOK, D_MODEL, D_MODEL, 0);
  // LN2 -> h2 (bf16)
  layernorm_bf16<<<NTOK, 256, 0, stream>>>(out, scale2, shift2, h2bf);
  // m = gelu(h2*W1^T + b1) (bf16)
  gemm_wmma_bf16<<<dim3(FFDIM / 128, NTOK / 128), 256, 0, stream>>>(
      h2bf, w1_b16, mbf, b1, nullptr, NTOK, FFDIM, D_MODEL, GF_GELU | GF_BF16);
  // out = x1 + m*W2^T + b2 (fp32 RMW of d_out)
  gemm_wmma_bf16<<<dim3(D_MODEL / 128, NTOK / 128), 256, 0, stream>>>(
      mbf, w2_b16, (void*)out, b2, out, NTOK, D_MODEL, FFDIM, 0);
}
